// RelationalDynamics_14955076125392
// MI455X (gfx1250) — matrix operand order repaired and checked by reading, not compiled, not run on hardware
//
#include <hip/hip_runtime.h>

// ---------------- problem constants ----------------
#define ZS   128
#define KOBJ 16
#define BB   2048
#define DIMX 256
#define BK   (BB * KOBJ)      // 32768 rows
#define NPAIR 120

typedef __attribute__((ext_vector_type(16))) _Float16 v16h;
typedef __attribute__((ext_vector_type(8)))  _Float16 v8h;
typedef __attribute__((ext_vector_type(4)))  _Float16 v4h;
typedef __attribute__((ext_vector_type(8)))  float    v8f;
typedef __attribute__((ext_vector_type(4)))  float    v4f;

union V16U { v16h v; v8h h[2]; };

// ---- WMMA wrapper: D = A(16x32 f16) * B(32x16 f16) + C(16x16 f32) ----
__device__ __forceinline__ v8f wmma_f16(v16h a, v16h b, v8f c) {
    return __builtin_amdgcn_wmma_f32_16x16x32_f16(
        /*neg_a=*/false, a, /*neg_b=*/false, b,
        /*c_mod=*/(short)0, c, /*reuse_a=*/false, /*reuse_b=*/false);
}

// A fragment (16x32, f16) from LDS, row-major, ld in halves.
// ISA layout: lane l (row = l&15, hi = l>>4): half[i] = X[row][k0 + (i>>3)*16 + hi*8 + (i&7)]
__device__ __forceinline__ v16h load_a_lds(const _Float16* base, int ld, int k0, int lane) {
    int row = lane & 15, hi = lane >> 4;
    const _Float16* p = base + row * ld + k0 + hi * 8;
    V16U u;
    u.h[0] = *(const v8h*)p;
    u.h[1] = *(const v8h*)(p + 16);
    return u.v;
}

// B fragment (32x16, f16) from global f16 weight stored row-major [N x K] (W[n][k]).
// ISA layout: lane l (col = l&15, hi = l>>4): half[i] = W[n0+col][k0 + (i>>3)*16 + hi*8 + (i&7)]
__device__ __forceinline__ v16h load_b_w(const _Float16* W, int ldk, int n0, int k0, int lane) {
    int col = lane & 15, hi = lane >> 4;
    const _Float16* p = W + (size_t)(n0 + col) * ldk + k0 + hi * 8;
    V16U u;
    u.h[0] = *(const v8h*)p;
    u.h[1] = *(const v8h*)(p + 16);
    return u.v;
}

__device__ __forceinline__ float sigm(float x) { return 1.0f / (1.0f + __expf(-x)); }
__device__ __forceinline__ float eluf(float x) { return x > 0.0f ? x : __expf(x) - 1.0f; }

// ---------------- kernel 0: fold + convert weights to f16 ----------------
__global__ void k_prep(const float* W_obj, const float* Wih, const float* Whh,
                       const float* We, const float* Wc,
                       const float* Wl, const float* Ws, const float* Wh,
                       _Float16* WobjH, _Float16* WihH, _Float16* WhhH,
                       _Float16* WeAH, _Float16* WeBH,
                       _Float16* WcHH, _Float16* WcEH,
                       _Float16* WlH, _Float16* WsH, _Float16* WhH) {
    int stride = gridDim.x * blockDim.x;
    int tid = blockIdx.x * blockDim.x + threadIdx.x;
    for (int i = tid; i < 256 * 128; i += stride) WobjH[i] = (_Float16)W_obj[i];
    for (int i = tid; i < 384 * 256; i += stride) WihH[i]  = (_Float16)Wih[i];
    for (int i = tid; i < 384 * 128; i += stride) WhhH[i]  = (_Float16)Whh[i];
    for (int i = tid; i < 512 * 128; i += stride) {
        int n = i >> 7, k = i & 127;
        WeAH[i] = (_Float16)(We[n * 512 + k]       + We[n * 512 + 128 + k]);
        WeBH[i] = (_Float16)(We[n * 512 + 256 + k] + We[n * 512 + 384 + k]);
    }
    for (int i = tid; i < 128 * 128; i += stride) {
        int n = i >> 7, k = i & 127;
        WcHH[i] = (_Float16)(Wc[n * 768 + k] + Wc[n * 768 + 128 + k]);
    }
    for (int i = tid; i < 128 * 512; i += stride) {
        int n = i >> 9, k = i & 511;
        WcEH[i] = (_Float16)Wc[n * 768 + 256 + k];
    }
    for (int i = tid; i < 128 * 128; i += stride) {
        WlH[i] = (_Float16)Wl[i];
        WsH[i] = (_Float16)Ws[i];
        WhH[i] = (_Float16)Wh[i];
    }
}

// ---------------- kernel 1: embedding + GRU step ----------------
// One wave per 16-row tile; 4 waves / block; 512 blocks -> 2048 tiles.
// LDS per wave: zT[16x136] f16, zE[16x264] f16, hT[16x136] f16.
#define LDZ 136
#define LDE 264
__global__ void k_embed_gru(const float* __restrict__ z, const float* __restrict__ h0,
                            const float* __restrict__ b_obj, const float* __restrict__ bih,
                            const float* __restrict__ bhh,
                            const _Float16* __restrict__ WobjH,
                            const _Float16* __restrict__ WihH,
                            const _Float16* __restrict__ WhhH,
                            _Float16* __restrict__ Hf16) {
    extern __shared__ _Float16 sm1[];
    const int lane = threadIdx.x & 31;
    const int wave = threadIdx.x >> 5;          // 0..3
    const int tile = blockIdx.x * 4 + wave;     // 0..2047
    const int row0 = tile * 16;
    _Float16* zT = sm1 + wave * 16 * (LDZ + LDE + LDZ);
    _Float16* zE = zT + 16 * LDZ;
    _Float16* hT = zE + 16 * LDE;

    // stage z & h0 tiles as f16 (vectorized: float4 -> half4)
    for (int idx = lane; idx < 16 * 32; idx += 32) {
        int r = idx >> 5, c4 = (idx & 31) * 4;
        v4f tz = *(const v4f*)&z [(size_t)(row0 + r) * ZS + c4];
        v4f th = *(const v4f*)&h0[(size_t)(row0 + r) * ZS + c4];
        v4h hz, hh;
#pragma unroll
        for (int j = 0; j < 4; ++j) { hz[j] = (_Float16)tz[j]; hh[j] = (_Float16)th[j]; }
        *(v4h*)&zT[r * LDZ + c4] = hz;
        *(v4h*)&hT[r * LDZ + c4] = hh;
    }
    __syncthreads();

    // z_embed = elu(z @ Wobj^T + b_obj) : [16 x 256]
    for (int nt = 0; nt < 16; ++nt) {
        v8f acc = {};
#pragma unroll
        for (int kk = 0; kk < ZS; kk += 32)
            acc = wmma_f16(load_a_lds(zT, LDZ, kk, lane),
                           load_b_w(WobjH, ZS, nt * 16, kk, lane), acc);
        int col = nt * 16 + (lane & 15);
        int rbase = (lane >> 4) * 8;
        float bias = b_obj[col];
#pragma unroll
        for (int v = 0; v < 8; ++v)
            zE[(rbase + v) * LDE + col] = (_Float16)eluf(acc[v] + bias);
    }
    __syncthreads();

    // GRU: per 16-col tile of the hidden state
    for (int t = 0; t < 8; ++t) {
        v8f ir = {}, iz = {}, in_ = {}, hr = {}, hz = {}, hn = {};
#pragma unroll
        for (int kk = 0; kk < DIMX; kk += 32) {
            v16h a = load_a_lds(zE, LDE, kk, lane);
            ir  = wmma_f16(a, load_b_w(WihH, DIMX, t * 16,        kk, lane), ir);
            iz  = wmma_f16(a, load_b_w(WihH, DIMX, 128 + t * 16,  kk, lane), iz);
            in_ = wmma_f16(a, load_b_w(WihH, DIMX, 256 + t * 16,  kk, lane), in_);
        }
#pragma unroll
        for (int kk = 0; kk < ZS; kk += 32) {
            v16h a = load_a_lds(hT, LDZ, kk, lane);
            hr = wmma_f16(a, load_b_w(WhhH, ZS, t * 16,       kk, lane), hr);
            hz = wmma_f16(a, load_b_w(WhhH, ZS, 128 + t * 16, kk, lane), hz);
            hn = wmma_f16(a, load_b_w(WhhH, ZS, 256 + t * 16, kk, lane), hn);
        }
        int col = t * 16 + (lane & 15);
        int rbase = (lane >> 4) * 8;
        float b_ir = bih[col], b_iz = bih[128 + col], b_in = bih[256 + col];
        float b_hr = bhh[col], b_hz = bhh[128 + col], b_hn = bhh[256 + col];
#pragma unroll
        for (int v = 0; v < 8; ++v) {
            float r = sigm(ir[v] + b_ir + hr[v] + b_hr);
            float u = sigm(iz[v] + b_iz + hz[v] + b_hz);
            float n = tanhf(in_[v] + b_in + r * (hn[v] + b_hn));
            float hp = (float)hT[(rbase + v) * LDZ + col];
            float ho = (1.0f - u) * n + u * hp;
            Hf16[(size_t)(row0 + rbase + v) * ZS + col] = (_Float16)ho;
        }
    }
}

// ---------------- kernel 2: pairwise relations + attention scatter ----------------
// One workgroup (256 threads, 8 waves) per batch.
#define LDH 136
__global__ void k_pairs(const _Float16* __restrict__ Hf16,
                        const _Float16* __restrict__ WeAH,
                        const _Float16* __restrict__ WeBH,
                        const float* __restrict__ be, const float* __restrict__ Wa,
                        const float* __restrict__ ba,
                        _Float16* __restrict__ Ef16) {
    extern __shared__ unsigned char sm2[];
    _Float16* Hb  = (_Float16*)sm2;                           // 16 x 136 f16
    float* G      = (float*)(sm2 + 16 * LDH * 2);             // 16 x 512 f32
    float* G2     = G  + 16 * 512;
    float* Eacc   = G2 + 16 * 512;
    const int b = blockIdx.x;
    const int tid = threadIdx.x, lane = tid & 31, wave = tid >> 5;

    for (int idx = tid; idx < 16 * 16; idx += 256) {          // 256 x v8h copies
        int r = idx >> 4, c8 = (idx & 15) * 8;
        *(v8h*)&Hb[r * LDH + c8] = *(const v8h*)&Hf16[(size_t)(b * 16 + r) * ZS + c8];
    }
    for (int idx = tid; idx < 16 * 512; idx += 256) Eacc[idx] = 0.0f;
    __syncthreads();

    // G = Hb @ WeA^T, G2 = Hb @ WeB^T  (M=16=KOBJ, N=512, K=128); 64 N-tiles over 8 waves
    for (int ti = wave; ti < 64; ti += 8) {
        const _Float16* W = (ti < 32) ? WeAH : WeBH;
        float* dst = (ti < 32) ? G : G2;
        int nt = ti & 31;
        v8f acc = {};
#pragma unroll
        for (int kk = 0; kk < ZS; kk += 32)
            acc = wmma_f16(load_a_lds(Hb, LDH, kk, lane),
                           load_b_w(W, ZS, nt * 16, kk, lane), acc);
        int col = nt * 16 + (lane & 15);
        int rbase = (lane >> 4) * 8;
#pragma unroll
        for (int v = 0; v < 8; ++v) dst[(rbase + v) * 512 + col] = acc[v];
    }
    __syncthreads();

    // pairs: eff = elu(G[i]+G2[j]+be); att = sigmoid(dot(eff,Wa)+ba); E[i]+=att*eff; E[j]+=...
    for (int p = wave; p < NPAIR; p += 8) {
        int i = 0, pp = p;
        while (pp >= (KOBJ - 1) - i) { pp -= (KOBJ - 1) - i; ++i; }
        int j = i + 1 + pp;
        float eff[16];
        float part = 0.0f;
        int c0 = lane * 16;
#pragma unroll
        for (int q = 0; q < 16; ++q) {
            int c = c0 + q;
            float x = eluf(G[i * 512 + c] + G2[j * 512 + c] + be[c]);
            eff[q] = x;
            part += x * Wa[c];
        }
#pragma unroll
        for (int off = 16; off > 0; off >>= 1) part += __shfl_xor(part, off, 32);
        float att = sigm(part + ba[0]);
#pragma unroll
        for (int q = 0; q < 16; ++q) {
            float w = att * eff[q];
            atomicAdd(&Eacc[i * 512 + c0 + q], w);
            atomicAdd(&Eacc[j * 512 + c0 + q], w);
        }
    }
    __syncthreads();
    for (int idx = tid; idx < 16 * 512; idx += 256)
        Ef16[(size_t)b * (16 * 512) + idx] = (_Float16)Eacc[idx];
}

// ---------------- kernel 3: rel projection + three heads ----------------
#define LDEV 520
#define LDR  136
__global__ void k_rel_heads(const _Float16* __restrict__ Hf16,
                            const _Float16* __restrict__ Ef16,
                            const _Float16* __restrict__ WcHH,
                            const _Float16* __restrict__ WcEH,
                            const _Float16* __restrict__ WlH,
                            const _Float16* __restrict__ WsH,
                            const _Float16* __restrict__ WhH,
                            const float* __restrict__ bc, const float* __restrict__ bl,
                            const float* __restrict__ bs, const float* __restrict__ bh,
                            float* __restrict__ out) {
    extern __shared__ _Float16 sm3[];
    const int lane = threadIdx.x & 31;
    const int wave = threadIdx.x >> 5;          // 0..3
    const int tile = blockIdx.x * 4 + wave;
    const int row0 = tile * 16;
    _Float16* Ht = sm3 + wave * 16 * (LDH + LDEV + LDR);
    _Float16* Et = Ht + 16 * LDH;
    _Float16* Rt = Et + 16 * LDEV;

    for (int idx = lane; idx < 16 * 16; idx += 32) {
        int r = idx >> 4, c8 = (idx & 15) * 8;
        *(v8h*)&Ht[r * LDH + c8] = *(const v8h*)&Hf16[(size_t)(row0 + r) * ZS + c8];
    }
    for (int idx = lane; idx < 16 * 64; idx += 32) {
        int r = idx >> 6, c8 = (idx & 63) * 8;
        *(v8h*)&Et[r * LDEV + c8] = *(const v8h*)&Ef16[(size_t)(row0 + r) * 512 + c8];
    }
    __syncthreads();

    // rel = H @ WcH^T + E @ WcE^T + bc : [16 x 128]
    for (int nt = 0; nt < 8; ++nt) {
        v8f acc = {};
#pragma unroll
        for (int kk = 0; kk < ZS; kk += 32)
            acc = wmma_f16(load_a_lds(Ht, LDH, kk, lane),
                           load_b_w(WcHH, ZS, nt * 16, kk, lane), acc);
#pragma unroll
        for (int kk = 0; kk < 512; kk += 32)
            acc = wmma_f16(load_a_lds(Et, LDEV, kk, lane),
                           load_b_w(WcEH, 512, nt * 16, kk, lane), acc);
        int col = nt * 16 + (lane & 15);
        int rbase = (lane >> 4) * 8;
        float bias = bc[col];
#pragma unroll
        for (int v = 0; v < 8; ++v)
            Rt[(rbase + v) * LDR + col] = (_Float16)(acc[v] + bias);
    }
    __syncthreads();

    // heads: loc, scale -> out1 [BK x 256] ; h_out -> out2 at offset BK*256
    for (int head = 0; head < 3; ++head) {
        const _Float16* W = (head == 0) ? WlH : (head == 1 ? WsH : WhH);
        const float* bb = (head == 0) ? bl : (head == 1 ? bs : bh);
        for (int nt = 0; nt < 8; ++nt) {
            v8f acc = {};
#pragma unroll
            for (int kk = 0; kk < ZS; kk += 32)
                acc = wmma_f16(load_a_lds(Rt, LDR, kk, lane),
                               load_b_w(W, ZS, nt * 16, kk, lane), acc);
            int col = nt * 16 + (lane & 15);
            int rbase = (lane >> 4) * 8;
            float bias = bb[col];
#pragma unroll
            for (int v = 0; v < 8; ++v) {
                float val = acc[v] + bias;
                int r = row0 + rbase + v;
                size_t o;
                if (head == 0)      o = (size_t)r * 256 + col;
                else if (head == 1) o = (size_t)r * 256 + 128 + col;
                else                o = (size_t)BK * 256 + (size_t)r * 128 + col;
                out[o] = val;
            }
        }
    }
}

// ---------------- launch ----------------
extern "C" void kernel_launch(void* const* d_in, const int* in_sizes, int n_in,
                              void* d_out, int out_size, void* d_ws, size_t ws_size,
                              hipStream_t stream) {
    const float* z     = (const float*)d_in[0];
    const float* h0    = (const float*)d_in[1];
    const float* W_obj = (const float*)d_in[2];  const float* b_obj = (const float*)d_in[3];
    const float* Wih   = (const float*)d_in[4];  const float* bih   = (const float*)d_in[5];
    const float* Whh   = (const float*)d_in[6];  const float* bhh   = (const float*)d_in[7];
    const float* We    = (const float*)d_in[8];  const float* be    = (const float*)d_in[9];
    const float* Wa    = (const float*)d_in[10]; const float* ba    = (const float*)d_in[11];
    const float* Wc    = (const float*)d_in[12]; const float* bc    = (const float*)d_in[13];
    const float* Wl    = (const float*)d_in[14]; const float* bl    = (const float*)d_in[15];
    const float* Ws    = (const float*)d_in[16]; const float* bs    = (const float*)d_in[17];
    const float* Wh    = (const float*)d_in[18]; const float* bh    = (const float*)d_in[19];
    float* out = (float*)d_out;

    // workspace layout (halves)
    _Float16* w = (_Float16*)d_ws;
    size_t o = 0;
    _Float16* WobjH = w + o; o += 256 * 128;
    _Float16* WihH  = w + o; o += 384 * 256;
    _Float16* WhhH  = w + o; o += 384 * 128;
    _Float16* WeAH  = w + o; o += 512 * 128;
    _Float16* WeBH  = w + o; o += 512 * 128;
    _Float16* WcHH  = w + o; o += 128 * 128;
    _Float16* WcEH  = w + o; o += 128 * 512;
    _Float16* WlH   = w + o; o += 128 * 128;
    _Float16* WsH   = w + o; o += 128 * 128;
    _Float16* WhH   = w + o; o += 128 * 128;
    _Float16* Hf16  = w + o; o += (size_t)BK * ZS;
    _Float16* Ef16  = w + o; o += (size_t)BK * 512;

    k_prep<<<256, 256, 0, stream>>>(W_obj, Wih, Whh, We, Wc, Wl, Ws, Wh,
                                    WobjH, WihH, WhhH, WeAH, WeBH,
                                    WcHH, WcEH, WlH, WsH, WhH);

    size_t sh1 = (size_t)4 * 16 * (LDZ + LDE + LDZ) * sizeof(_Float16);   // ~68.6 KB
    k_embed_gru<<<512, 128, sh1, stream>>>(z, h0, b_obj, bih, bhh,
                                           WobjH, WihH, WhhH, Hf16);

    size_t sh2 = (size_t)16 * LDH * sizeof(_Float16) + (size_t)3 * 16 * 512 * sizeof(float); // ~102.7 KB
    k_pairs<<<BB, 256, sh2, stream>>>(Hf16, WeAH, WeBH, be, Wa, ba, Ef16);

    size_t sh3 = (size_t)4 * 16 * (LDH + LDEV + LDR) * sizeof(_Float16);  // ~101.4 KB
    k_rel_heads<<<512, 128, sh3, stream>>>(Hf16, Ef16, WcHH, WcEH, WlH, WsH, WhH,
                                           bc, bl, bs, bh, out);
}
